// GraphConditioner_13391708029190
// MI455X (gfx1250) — matrix operand order repaired and checked by reading, not compiled, not run on hardware
//
#include <hip/hip_runtime.h>
#include <hip/hip_bf16.h>

// ---------- types for WMMA ----------
typedef __attribute__((ext_vector_type(16))) __bf16 v16bf;
typedef __attribute__((ext_vector_type(8)))  __bf16 v8bf;
typedef __attribute__((ext_vector_type(8)))  float  v8f;

union V16 { v16bf v; v8bf h[2]; };

// ---------- helpers ----------
__device__ __forceinline__ __bf16 f2bf(float f) {
    union { float f; unsigned u; } c; c.f = f;
    unsigned r = c.u + 0x7FFFu + ((c.u >> 16) & 1u);   // round-to-nearest-even
    union { unsigned short s; __bf16 b; } o; o.s = (unsigned short)(r >> 16);
    return o.b;
}
// order-preserving float <-> uint key (for atomic max on floats)
__device__ __forceinline__ unsigned f2key(float f) {
    union { float f; unsigned u; } c; c.f = f;
    return (c.u & 0x80000000u) ? ~c.u : (c.u | 0x80000000u);
}
__device__ __forceinline__ float key2f(unsigned u) {
    union { unsigned u; float f; } c;
    c.u = (u & 0x80000000u) ? (u & 0x7FFFFFFFu) : ~u;
    return c.f;
}
#define KEY_NEG_INF 0x007FFFFFu   /* f2key(-inf) */
#define NEG_SLOPE 0.2f
#define SM_EPS 1e-16f

// async memory->LDS 16B copy (ASYNCcnt-tracked; no VGPR data round-trip).
// Generic pointers into LDS carry the LDS offset in their low 32 bits (ISA 10.2).
__device__ __forceinline__ void async_copy_b128(void* lds_ptr, const void* gptr) {
    unsigned loff = (unsigned)(size_t)lds_ptr;
    asm volatile("global_load_async_to_lds_b128 %0, %1, off"
                 :: "v"(loff), "v"(gptr) : "memory");
}
__device__ __forceinline__ void wait_async0() {
    asm volatile("s_wait_asynccnt 0x0" ::: "memory");
}

// ---------- tiny utility kernels ----------
__global__ void fill_f32(float* p, int n, float v) {
    int i = blockIdx.x * blockDim.x + threadIdx.x;
    if (i < n) p[i] = v;
}
__global__ void fill_u32(unsigned* p, int n, unsigned v) {
    int i = blockIdx.x * blockDim.x + threadIdx.x;
    if (i < n) p[i] = v;
}
__global__ void cvt_f32_bf16(const float* __restrict__ in, __bf16* __restrict__ out, int n) {
    int i = blockIdx.x * blockDim.x + threadIdx.x;
    if (i < n) out[i] = f2bf(in[i]);
}
// W [K, Nout] fp32 -> Wt [Nout, K] bf16
__global__ void transpose_w_bf16(const float* __restrict__ W, __bf16* __restrict__ Wt,
                                 int K, int Nout) {
    int idx = blockIdx.x * blockDim.x + threadIdx.x;
    if (idx >= K * Nout) return;
    int k = idx / Nout, n = idx % Nout;
    Wt[(size_t)n * K + k] = f2bf(W[(size_t)k * Nout + n]);
}

// ---------- WMMA bf16 GEMM:  C[M,Nout] = A[M,K](bf16) * Bt[Nout,K](bf16) ----------
// 128x128x32 block, 8 waves (4x2), 32x64 per wave = 2x4 WMMA tiles.
// Double-buffered LDS staged with global_load_async_to_lds_b128 so tile k+1's
// HBM traffic overlaps tile k's 8 v_wmma ops; s_wait_asynccnt + barrier per step.
#define BM 128
#define BN 128
#define BK 32
#define LDSS 40   // padded K-stride in elements (80 bytes, 16B aligned)

__global__ __launch_bounds__(256)
void gemm_bf16_wmma(const __bf16* __restrict__ A, const __bf16* __restrict__ Bt,
                    float* __restrict__ C, int M, int Nout, int K) {
    __shared__ __attribute__((aligned(16))) __bf16 sA[2][BM * LDSS];
    __shared__ __attribute__((aligned(16))) __bf16 sB[2][BN * LDSS];

    const int tid   = threadIdx.x;
    const int wave  = tid >> 5;
    const int lane  = tid & 31;
    const int lhalf = lane >> 4;   // 0/1
    const int l15   = lane & 15;

    const int nblk = Nout / BN;
    const int bm = blockIdx.x / nblk;
    const int bn = blockIdx.x % nblk;

    const int wm = wave >> 1;      // 0..3  (rows)
    const int wn = wave & 1;       // 0..1  (cols)

    const int lrow = tid >> 2;            // 0..63
    const int lcol = (tid & 3) * 8;       // element offset in K-tile

    // Source rows. OOB A rows are clamped: A-tile row r only ever feeds C-tile
    // row r, and the C store is row-guarded, so clamped (garbage) rows are inert.
    int ga0 = bm * BM + lrow;        if (ga0 >= M) ga0 = M - 1;
    int ga1 = bm * BM + lrow + 64;   if (ga1 >= M) ga1 = M - 1;
    const int gb0 = bn * BN + lrow;        // Nout multiple of BN: always in range
    const int gb1 = bn * BN + lrow + 64;

    auto stage = [&](int buf, int k0) {
        async_copy_b128(&sA[buf][lrow * LDSS + lcol],        A  + (size_t)ga0 * K + k0 + lcol);
        async_copy_b128(&sA[buf][(lrow + 64) * LDSS + lcol], A  + (size_t)ga1 * K + k0 + lcol);
        async_copy_b128(&sB[buf][lrow * LDSS + lcol],        Bt + (size_t)gb0 * K + k0 + lcol);
        async_copy_b128(&sB[buf][(lrow + 64) * LDSS + lcol], Bt + (size_t)gb1 * K + k0 + lcol);
    };

    v8f acc[2][4] = {};

    stage(0, 0);
    wait_async0();
    __syncthreads();

    const int nk = K / BK;
    for (int kt = 0; kt < nk; ++kt) {
        const int cur = kt & 1;
        if (kt + 1 < nk) stage(cur ^ 1, (kt + 1) * BK);   // async prefetch next tile

        // ---- load fragments per documented 16x32 bf16 layouts ----
        V16 af[2], bf[4];
        #pragma unroll
        for (int tm = 0; tm < 2; ++tm) {
            const __bf16* p = &sA[cur][(wm * 32 + tm * 16 + l15) * LDSS + lhalf * 8];
            af[tm].h[0] = *(const v8bf*)p;          // K = 8*lhalf + 0..7
            af[tm].h[1] = *(const v8bf*)(p + 16);   // K = 16 + 8*lhalf + 0..7
        }
        #pragma unroll
        for (int tn = 0; tn < 4; ++tn) {
            const __bf16* p = &sB[cur][(wn * 64 + tn * 16 + l15) * LDSS + lhalf * 8];
            bf[tn].h[0] = *(const v8bf*)p;          // K = 8*lhalf + 0..7
            bf[tn].h[1] = *(const v8bf*)(p + 16);   // K = 16 + 8*lhalf + 0..7
        }
        #pragma unroll
        for (int tm = 0; tm < 2; ++tm)
            #pragma unroll
            for (int tn = 0; tn < 4; ++tn)
                acc[tm][tn] = __builtin_amdgcn_wmma_f32_16x16x32_bf16(
                    false, af[tm].v, false, bf[tn].v,
                    (short)0, acc[tm][tn], false, false);

        wait_async0();       // next tile landed in LDS
        __syncthreads();     // all waves done reading current tile
    }

    // ---- write C per C/D layout: lane l -> N=l&15, VGPR v -> M = v + 8*(l>>4) ----
    const int row0 = bm * BM + wm * 32;
    const int col0 = bn * BN + wn * 64;
    #pragma unroll
    for (int tm = 0; tm < 2; ++tm)
        #pragma unroll
        for (int tn = 0; tn < 4; ++tn) {
            int col = col0 + tn * 16 + l15;
            #pragma unroll
            for (int v = 0; v < 8; ++v) {
                int row = row0 + tm * 16 + v + 8 * lhalf;
                if (row < M) C[(size_t)row * Nout + col] = acc[tm][tn][v];
            }
        }
}

// ---------- per-node attention coefficients: a_src = h.att_s, a_dst = h.att_d ----------
__global__ void attn_coeff_kernel(const float* __restrict__ h, int N, int F,
                                  const float* __restrict__ att_s,
                                  const float* __restrict__ att_d,
                                  float* __restrict__ a_src, float* __restrict__ a_dst) {
    int w = (blockIdx.x * blockDim.x + threadIdx.x) >> 5;
    int lane = threadIdx.x & 31;
    if (w >= N) return;
    const float* row = h + (size_t)w * F;
    float ss = 0.f, sd = 0.f;
    for (int f = lane; f < F; f += 32) {
        float v = row[f];
        ss += v * att_s[f];
        sd += v * att_d[f];
    }
    for (int m = 16; m; m >>= 1) { ss += __shfl_xor(ss, m, 32); sd += __shfl_xor(sd, m, 32); }
    if (lane == 0) { a_src[w] = ss; a_dst[w] = sd; }
}

// ---------- edge pass 1: segment max of leaky-relu scores ----------
__global__ void edge_max_kernel(const long long* __restrict__ src,
                                const long long* __restrict__ dst,
                                const float* __restrict__ a_src,
                                const float* __restrict__ a_dst,
                                unsigned* __restrict__ m_u, int E, int N) {
    int e = blockIdx.x * blockDim.x + threadIdx.x;
    if (e >= E + N) return;
    int s, d;
    if (e < E) { s = (int)src[e]; d = (int)dst[e]; } else { s = d = e - E; }
    float sc = a_src[s] + a_dst[d];
    sc = sc > 0.f ? sc : NEG_SLOPE * sc;
    atomicMax(m_u + d, f2key(sc));
}

// ---------- edge pass 2: w = exp(score - max); scatter w*h[src] and w ----------
__global__ void edge_accum_kernel(const long long* __restrict__ src,
                                  const long long* __restrict__ dst,
                                  const float* __restrict__ a_src,
                                  const float* __restrict__ a_dst,
                                  const unsigned* __restrict__ m_u,
                                  const float* __restrict__ h,
                                  float* __restrict__ acc, float* __restrict__ denom,
                                  int E, int N, int F) {
    int w = (blockIdx.x * blockDim.x + threadIdx.x) >> 5;
    int lane = threadIdx.x & 31;
    if (w >= E + N) return;
    int s, d;
    if (w < E) { s = (int)src[w]; d = (int)dst[w]; } else { s = d = w - E; }
    float sc = a_src[s] + a_dst[d];
    sc = sc > 0.f ? sc : NEG_SLOPE * sc;
    float ew = __expf(sc - key2f(m_u[d]));
    if (lane == 0) atomicAdd(denom + d, ew);
    const float* hs = h + (size_t)s * F;
    float* ad = acc + (size_t)d * F;
    for (int f = lane; f < F; f += 32)
        atomicAdd(ad + f, ew * hs[f]);
}

// ---------- normalize + bias + relu; write fp32 and bf16 copies ----------
__global__ void normalize_kernel(const float* __restrict__ acc,
                                 const float* __restrict__ denom,
                                 const float* __restrict__ bias,
                                 float* __restrict__ out_f32,
                                 __bf16* __restrict__ out_bf, int N, int F) {
    int idx = blockIdx.x * blockDim.x + threadIdx.x;
    if (idx >= N * F) return;
    int i = idx / F, f = idx % F;
    float v = acc[idx] / (denom[i] + SM_EPS) + bias[f];
    v = v > 0.f ? v : 0.f;
    out_f32[idx] = v;
    out_bf[idx]  = f2bf(v);
}

// ---------- global attention ----------
__global__ void gate_kernel(const float* __restrict__ h, int N, int F,
                            const float* __restrict__ gw, const float* __restrict__ gb,
                            float* __restrict__ gate) {
    int w = (blockIdx.x * blockDim.x + threadIdx.x) >> 5;
    int lane = threadIdx.x & 31;
    if (w >= N) return;
    const float* row = h + (size_t)w * F;
    float s = 0.f;
    for (int f = lane; f < F; f += 32) s += row[f] * gw[f];
    for (int m = 16; m; m >>= 1) s += __shfl_xor(s, m, 32);
    if (lane == 0) gate[w] = s + gb[0];
}
__global__ void gmax_kernel(const float* __restrict__ gate, unsigned* __restrict__ gmax_u, int N) {
    int i = blockIdx.x * blockDim.x + threadIdx.x;
    if (i < N) atomicMax(gmax_u, f2key(gate[i]));
}
__global__ void gsum_kernel(const float* __restrict__ gate,
                            const unsigned* __restrict__ gmax_u,
                            const float* __restrict__ h,
                            float* __restrict__ out_acc, float* __restrict__ wsum,
                            int N, int F) {
    const int CHUNK = 64;
    int f = threadIdx.x;               // blockDim == F == 256
    int n0 = blockIdx.x * CHUNK;
    int n1 = n0 + CHUNK; if (n1 > N) n1 = N;
    float gm = key2f(*gmax_u);
    float pf = 0.f, pw = 0.f;
    for (int i = n0; i < n1; ++i) {
        float w = __expf(gate[i] - gm);
        pf += w * h[(size_t)i * F + f];
        pw += w;
    }
    atomicAdd(out_acc + f, pf);
    if (f == 0) atomicAdd(wsum, pw);
}
__global__ void finalize_kernel(const float* __restrict__ out_acc,
                                const float* __restrict__ wsum,
                                float* __restrict__ out, int F) {
    int f = threadIdx.x;
    if (f < F) out[f] = out_acc[f] / wsum[0];
}

// ---------- host launch ----------
extern "C" void kernel_launch(void* const* d_in, const int* in_sizes, int n_in,
                              void* d_out, int out_size, void* d_ws, size_t ws_size,
                              hipStream_t stream) {
    const float*     x   = (const float*)d_in[0];
    const long long* ei  = (const long long*)d_in[1];
    const float*     W1  = (const float*)d_in[2];
    const float*     b1  = (const float*)d_in[3];
    const float*     as1 = (const float*)d_in[4];
    const float*     ad1 = (const float*)d_in[5];
    const float*     W2  = (const float*)d_in[6];
    const float*     b2  = (const float*)d_in[7];
    const float*     as2 = (const float*)d_in[8];
    const float*     ad2 = (const float*)d_in[9];
    const float*     gw  = (const float*)d_in[10];
    const float*     gb  = (const float*)d_in[11];

    const int D = 768, F = 256;
    const int N = in_sizes[0] / D;
    const int E = in_sizes[1] / 2;
    const long long* src = ei;
    const long long* dst = ei + E;

    // carve workspace
    char* wp = (char*)d_ws;
    auto carve = [&](size_t bytes) { void* r = (void*)wp; wp += (bytes + 255) & ~(size_t)255; return r; };
    __bf16* x_bf   = (__bf16*)carve((size_t)N * D * 2);
    __bf16* w1t    = (__bf16*)carve((size_t)D * F * 2);
    __bf16* w2t    = (__bf16*)carve((size_t)F * F * 2);
    float*  h_lin  = (float*) carve((size_t)N * F * 4);   // GEMM output (pre-aggregation features)
    float*  agg    = (float*) carve((size_t)N * F * 4);   // scatter accumulator
    float*  h_f32  = (float*) carve((size_t)N * F * 4);   // layer output fp32
    __bf16* h_bf   = (__bf16*)carve((size_t)N * F * 2);   // layer output bf16 (next GEMM input)
    float*  a_src  = (float*) carve((size_t)N * 4);
    float*  a_dst  = (float*) carve((size_t)N * 4);
    float*  denom  = (float*) carve((size_t)N * 4);
    unsigned* m_u  = (unsigned*)carve((size_t)N * 4);
    float*  gate   = (float*) carve((size_t)N * 4);
    float*  outacc = (float*) carve(256 * 4);
    float*  wsum   = (float*) carve(4);
    unsigned* gmax = (unsigned*)carve(4);

    const int T = 256;
    auto cdiv = [](long long a, long long b) { return (int)((a + b - 1) / b); };

    // pack inputs to bf16 (W transposed to [Nout][K])
    cvt_f32_bf16<<<cdiv((long long)N * D, T), T, 0, stream>>>(x, x_bf, N * D);
    transpose_w_bf16<<<cdiv((long long)D * F, T), T, 0, stream>>>(W1, w1t, D, F);
    transpose_w_bf16<<<cdiv((long long)F * F, T), T, 0, stream>>>(W2, w2t, F, F);

    const int mblk = cdiv(N, BM);
    const int nblk = F / BN;
    const int EW_blocks = cdiv((long long)(E + N), 8);    // 8 waves/block, 1 wave/edge
    const int NW_blocks = cdiv(N, 8);

    for (int layer = 0; layer < 2; ++layer) {
        // GEMM: features = input @ W
        if (layer == 0)
            gemm_bf16_wmma<<<mblk * nblk, T, 0, stream>>>(x_bf, w1t, h_lin, N, F, D);
        else
            gemm_bf16_wmma<<<mblk * nblk, T, 0, stream>>>(h_bf, w2t, h_lin, N, F, F);

        attn_coeff_kernel<<<NW_blocks, T, 0, stream>>>(
            h_lin, N, F, layer ? as2 : as1, layer ? ad2 : ad1, a_src, a_dst);

        fill_u32<<<cdiv(N, T), T, 0, stream>>>(m_u, N, KEY_NEG_INF);
        fill_f32<<<cdiv(N, T), T, 0, stream>>>(denom, N, 0.f);
        fill_f32<<<cdiv((long long)N * F, T), T, 0, stream>>>(agg, N * F, 0.f);

        edge_max_kernel<<<cdiv(E + N, T), T, 0, stream>>>(src, dst, a_src, a_dst, m_u, E, N);
        edge_accum_kernel<<<EW_blocks, T, 0, stream>>>(src, dst, a_src, a_dst, m_u,
                                                       h_lin, agg, denom, E, N, F);
        normalize_kernel<<<cdiv((long long)N * F, T), T, 0, stream>>>(
            agg, denom, layer ? b2 : b1, h_f32, h_bf, N, F);
    }

    // global attention over h_f32
    gate_kernel<<<NW_blocks, T, 0, stream>>>(h_f32, N, F, gw, gb, gate);
    fill_u32<<<1, 1, 0, stream>>>(gmax, 1, KEY_NEG_INF);
    gmax_kernel<<<cdiv(N, T), T, 0, stream>>>(gate, gmax, N);
    fill_f32<<<1, 256, 0, stream>>>(outacc, 256, 0.f);
    fill_f32<<<1, 1, 0, stream>>>(wsum, 1, 0.f);
    gsum_kernel<<<cdiv(N, 64), 256, 0, stream>>>(gate, gmax, h_f32, outacc, wsum, N, F);
    finalize_kernel<<<1, 256, 0, stream>>>(outacc, wsum, (float*)d_out, F);
}
